// LSTMFC_9079560863815
// MI455X (gfx1250) — compile-verified, matrix operand order checked by reading, not run on hardware
//
// LSTM with done-resets, MI455X (gfx1250).
// Phase 1: xz = bf16(inputs) @ bf16(wx) precomputed as one big WMMA GEMM (parallel).
// Phase 2: persistent 16-WG kernel; per step h@wh via v_wmma_f32_16x16x32_bf16 with
//          wh held in LDS (320KB/WGP); h staged via GLOBAL_LOAD_ASYNC_TO_LDS_B128
//          (ASYNCcnt); xz stream prefetched with global_prefetch_b8; global spin
//          barrier per step. Workspace requirement: ~273 MB.

#include <hip/hip_runtime.h>

typedef __attribute__((ext_vector_type(16))) __bf16 v16bf;
typedef __attribute__((ext_vector_type(8)))  float  v8f;
typedef unsigned short u16b;

union Frag {
    v16bf v;
    uint4 q[2];
};

#define NWG 16          // persistent workgroups in recurrent kernel
#define BB 64
#define TT 1024
#define DD 512
#define HH 512
#define G4H 2048
#define HPAD 520        // padded LDS row stride (halves) to spread banks

__device__ __forceinline__ u16b f2bf(float f) {
    unsigned int u = __float_as_uint(f);
    unsigned int r = u + 0x7fffu + ((u >> 16) & 1u);   // round-to-nearest-even
    return (u16b)(r >> 16);
}
__device__ __forceinline__ float bf2f(u16b s) {
    return __uint_as_float(((unsigned int)s) << 16);
}
__device__ __forceinline__ float sigm(float x) { return 1.0f / (1.0f + __expf(-x)); }
__device__ __forceinline__ float tanh_f(float x) {
    float e = __expf(-2.0f * fabsf(x));
    float r = (1.0f - e) / (1.0f + e);
    return x < 0.0f ? -r : r;
}
// 16-bit A/B fragment element mapping (ISA 7.12.2): half j -> local K
__device__ __forceinline__ int klocal(int j) {
    int i = j >> 1;
    return (i < 4 ? 2 * i : 16 + 2 * (i - 4)) + (j & 1);
}

// Async DMA of 16 bytes/lane from global into LDS (CDNA5, tracked by ASYNCcnt).
// Generic pointers into LDS hold the LDS byte offset in their low 32 bits
// (flat-address aperture rule, ISA 10.2), which is what VDST wants.
__device__ __forceinline__ void async_copy_b128(void* lds_dst, const void* gsrc) {
    unsigned int lds_off = (unsigned int)(size_t)lds_dst;
    asm volatile("global_load_async_to_lds_b128 %0, %1, off"
                 :: "v"(lds_off), "v"(gsrc)
                 : "memory");
}
__device__ __forceinline__ void wait_async0() {
    asm volatile("s_wait_asynccnt 0x0" ::: "memory");
}

// ---------------- prep: pack wx (fp32) -> bf16 B-fragment order ----------------
// layout: [nt(128)][kc(16)][lane(32)][jj(16)]
__global__ __launch_bounds__(256) void pack_wx(const float* __restrict__ wx, u16b* __restrict__ wxP) {
    int idx = blockIdx.x * 256 + threadIdx.x;           // 4096 blocks -> 1,048,576
    int jj = idx & 15, lane = (idx >> 4) & 31, kc = (idx >> 9) & 15, nt = idx >> 13;
    int n = nt * 16 + (lane & 15);
    int K = kc * 32 + (lane >> 4) * 8 + klocal(jj);
    wxP[idx] = f2bf(wx[K * G4H + n]);
}

// pack wh -> per-WG-slice fragment order: [w(16)][jl(2)][g(4)][kc(16)][lane(32)][jj(16)]
__global__ __launch_bounds__(256) void pack_wh(const float* __restrict__ wh, u16b* __restrict__ whP) {
    int idx = blockIdx.x * 256 + threadIdx.x;           // 4096 blocks -> 1,048,576
    int jj = idx & 15, lane = (idx >> 4) & 31, kc = (idx >> 9) & 15;
    int g = (idx >> 13) & 3, jl = (idx >> 15) & 1, w = idx >> 16;
    int col = g * HH + (w * 2 + jl) * 16 + (lane & 15);
    int K = kc * 32 + (lane >> 4) * 8 + klocal(jj);
    whP[idx] = f2bf(wh[K * G4H + col]);
}

// init: hbuf(parity0) = bf16(h0 * (1-done[:,0])), zero barrier counters
__global__ __launch_bounds__(256) void lstm_init(const float* __restrict__ state,
                                                 const float* __restrict__ done,
                                                 u16b* __restrict__ hbuf, int* __restrict__ ctr) {
    int e = blockIdx.x * 256 + threadIdx.x;             // 128 blocks -> 32768
    if (e < BB * HH) {
        int b = e >> 9, col = e & 511;
        float h0 = state[b * (2 * HH) + HH + col] * (1.0f - done[b * TT]);
        hbuf[e] = f2bf(h0);
    }
    if (e < TT) ctr[e] = 0;
}

// ---------------- phase 1: xz[bt, 4H] = X[bt, D] @ wx ----------------
__global__ __launch_bounds__(256) void lstm_xgemm(const float* __restrict__ X,
                                                  const u16b* __restrict__ wxP,
                                                  u16b* __restrict__ xz) {
    __shared__ u16b xs[16 * HPAD];                      // 16 rows x 512 cols bf16, padded
    int tid = threadIdx.x;
    long rowBase = (long)blockIdx.x * 16;
    for (int e = tid; e < 16 * DD; e += 256) {
        int r = e >> 9, col = e & 511;
        xs[r * HPAD + col] = f2bf(X[(rowBase + r) * DD + col]);
    }
    __syncthreads();

    int wave = tid >> 5, lane = tid & 31;
    int n = lane & 15, half = lane >> 4;
    int mbase = half * 8, koff = half * 8, ml = lane & 15;

    v8f acc[16] = {};
    for (int kc = 0; kc < 16; ++kc) {
        Frag a;
        const u16b* hr = xs + ml * HPAD + kc * 32 + koff;
        a.q[0] = *(const uint4*)(hr);
        a.q[1] = *(const uint4*)(hr + 16);
#pragma unroll
        for (int i = 0; i < 16; ++i) {
            int nt = wave + (i << 3);
            const u16b* bp = wxP + (((size_t)nt * 16 + kc) * 32 + lane) * 16;
            Frag bfr;
            bfr.q[0] = ((const uint4*)bp)[0];
            bfr.q[1] = ((const uint4*)bp)[1];
            acc[i] = __builtin_amdgcn_wmma_f32_16x16x32_bf16(
                false, a.v, false, bfr.v, (short)0, acc[i], false, false);
        }
    }
#pragma unroll
    for (int i = 0; i < 16; ++i) {
        int nt = wave + (i << 3);
#pragma unroll
        for (int r = 0; r < 8; ++r) {
            long row = rowBase + mbase + r;
            xz[row * G4H + nt * 16 + n] = f2bf(acc[i][r]);
        }
    }
}

// ---------------- phase 2: persistent recurrent kernel ----------------
__global__ __launch_bounds__(256) void lstm_rec(const float* __restrict__ done,
                                                const float* __restrict__ state,
                                                const float* __restrict__ bias,
                                                const u16b* __restrict__ whP,
                                                const u16b* __restrict__ xz,
                                                u16b* __restrict__ hbuf,
                                                int* __restrict__ ctr,
                                                float* __restrict__ out) {
    extern __shared__ char smemraw[];
    u16b* whs = (u16b*)smemraw;                               // 65536 halves = 128 KB
    u16b* hs = (u16b*)(smemraw + 131072);                     // 64 x 520 halves = 66560 B
    float* bsh = (float*)(smemraw + 131072 + 66560);          // 2048 floats
    float* dsh = bsh + G4H;                                   // 64 floats

    int tid = threadIdx.x, w = blockIdx.x;
    int wave = tid >> 5, lane = tid & 31;
    int m = wave & 3, jl = wave >> 2;
    int n = lane & 15, half = lane >> 4;
    int mbase = half * 8, koff = half * 8, ml = lane & 15;
    int hc = (w * 2 + jl) * 16 + n;

    // preload this WG's wh slice (128 KB) into LDS via async DMA + bias
    for (int e = tid; e < 8192; e += 256)
        async_copy_b128((uint4*)whs + e, (const uint4*)(whP + (size_t)w * 65536) + e);
    for (int e = tid; e < G4H; e += 256) bsh[e] = bias[e];

    float c[8], hlast[8];
#pragma unroll
    for (int r = 0; r < 8; ++r) {
        int b = 16 * m + mbase + r;
        c[r] = state[b * (2 * HH) + hc];
        hlast[r] = state[b * (2 * HH) + HH + hc];
    }
    wait_async0();
    __syncthreads();

    for (int t = 0; t < TT; ++t) {
        // stage h(t) (already scaled by 1-done[t]) global -> LDS via async DMA
        {
            const uint4* src = (const uint4*)(hbuf + (size_t)(t & 1) * (BB * HH));
            for (int e = tid; e < 4096; e += 256) {
                int row = e >> 6, cq = e & 63;
                async_copy_b128(hs + row * HPAD + cq * 8, src + e);
            }
        }
        if (tid < BB) dsh[tid] = done[tid * TT + t];
        // prefetch this step's xz gate slabs (consumed after the GEMM below)
        {
            size_t xo0 = ((size_t)(16 * m + mbase) * TT + t) * G4H + hc;
#pragma unroll
            for (int g = 0; g < 4; ++g)
                __builtin_prefetch(xz + xo0 + g * HH, 0, 1);
        }
        wait_async0();
        __syncthreads();

        v8f acc[4] = {};
        for (int kc = 0; kc < 16; ++kc) {
            Frag a;
            const u16b* hr = hs + (16 * m + ml) * HPAD + kc * 32 + koff;
            a.q[0] = *(const uint4*)(hr);
            a.q[1] = *(const uint4*)(hr + 16);
#pragma unroll
            for (int g = 0; g < 4; ++g) {
                const u16b* bp = whs + ((((jl * 4 + g) * 16) + kc) * 32 + lane) * 16;
                Frag bfr;
                bfr.q[0] = ((const uint4*)bp)[0];
                bfr.q[1] = ((const uint4*)bp)[1];
                acc[g] = __builtin_amdgcn_wmma_f32_16x16x32_bf16(
                    false, a.v, false, bfr.v, (short)0, acc[g], false, false);
            }
        }

#pragma unroll
        for (int r = 0; r < 8; ++r) {
            int b = 16 * m + mbase + r;
            float cc = c[r] * (1.0f - dsh[b]);
            size_t xo = ((size_t)b * TT + t) * G4H + hc;
            float iv = sigm(acc[0][r] + bf2f(xz[xo]) + bsh[hc]);
            float fv = sigm(acc[1][r] + bf2f(xz[xo + HH]) + bsh[HH + hc]);
            float ov = sigm(acc[2][r] + bf2f(xz[xo + 2 * HH]) + bsh[2 * HH + hc]);
            float uv = tanh_f(acc[3][r] + bf2f(xz[xo + 3 * HH]) + bsh[3 * HH + hc]);
            cc = fv * cc + iv * uv;
            c[r] = cc;
            float hv = ov * tanh_f(cc);
            hlast[r] = hv;
            out[((size_t)b * TT + t) * HH + hc] = hv;
            // fold next step's done-reset into the stored h
            float sc = (t + 1 < TT) ? (1.0f - done[b * TT + (t + 1)]) : 1.0f;
            hbuf[(size_t)((t + 1) & 1) * (BB * HH) + b * HH + hc] = f2bf(hv * sc);
        }
        __syncthreads();
        // grid-wide barrier: release h(t+1) to all WGs
        if (tid == 0) {
            __hip_atomic_fetch_add(&ctr[t], 1, __ATOMIC_ACQ_REL, __HIP_MEMORY_SCOPE_AGENT);
            while (__hip_atomic_load(&ctr[t], __ATOMIC_ACQUIRE, __HIP_MEMORY_SCOPE_AGENT) < NWG)
                __builtin_amdgcn_s_sleep(1);
        }
        __syncthreads();
    }

    // final state_out = [c, h]  appended after outputs
#pragma unroll
    for (int r = 0; r < 8; ++r) {
        int b = 16 * m + mbase + r;
        out[(size_t)BB * TT * HH + b * (2 * HH) + hc] = c[r];
        out[(size_t)BB * TT * HH + b * (2 * HH) + HH + hc] = hlast[r];
    }
}

extern "C" void kernel_launch(void* const* d_in, const int* in_sizes, int n_in,
                              void* d_out, int out_size, void* d_ws, size_t ws_size,
                              hipStream_t stream) {
    const float* inputs = (const float*)d_in[0];
    const float* done = (const float*)d_in[1];
    const float* state = (const float*)d_in[2];
    const float* wx = (const float*)d_in[3];
    const float* wh = (const float*)d_in[4];
    const float* bias = (const float*)d_in[5];
    float* out = (float*)d_out;

    char* ws = (char*)d_ws;
    u16b* wxP = (u16b*)(ws);                                        // 2 MB
    u16b* whP = (u16b*)(ws + (size_t)2 * 1024 * 1024);              // 2 MB
    u16b* xz = (u16b*)(ws + (size_t)4 * 1024 * 1024);               // 256 MB
    u16b* hbuf = (u16b*)(ws + (size_t)4 * 1024 * 1024 + 268435456u);// 128 KB (double buffer)
    int* ctr = (int*)(ws + (size_t)4 * 1024 * 1024 + 268435456u + 131072u); // 4 KB

    pack_wx<<<4096, 256, 0, stream>>>(wx, wxP);
    pack_wh<<<4096, 256, 0, stream>>>(wh, whP);
    lstm_init<<<128, 256, 0, stream>>>(state, done, hbuf, ctr);
    lstm_xgemm<<<4096, 256, 0, stream>>>(inputs, wxP, xz);

    size_t smem = 131072 + 66560 + G4H * 4 + BB * 4;  // ~206 KB of the 320 KB WGP LDS
    lstm_rec<<<NWG, 256, smem, stream>>>(done, state, bias, whP, xz, hbuf, ctr, out);
}
